// MambaBlock_55825984913750
// MI455X (gfx1250) — compile-verified, matrix operand order checked by reading, not run on hardware
//
#include <hip/hip_runtime.h>
#include <hip/hip_bf16.h>

// ---------------------------------------------------------------------------
// Mamba block forward for MI455X (gfx1250, wave32, WMMA + async LDS copies).
// GEMMs: bf16 operands staged global->LDS with GLOBAL_LOAD_ASYNC_TO_LDS_B128
// (ASYNCcnt-tracked, double buffered, branchless steady-state), fp32
// accumulate via v_wmma_f32_16x16x32_bf16 (8 WMMA per wave per K-chunk).
// Producers emit bf16 activation copies so the GEMM loop has no conversions.
// ---------------------------------------------------------------------------

typedef __attribute__((ext_vector_type(16))) __bf16 v16bf;
typedef __attribute__((ext_vector_type(8)))  __bf16 v8bf;
typedef __attribute__((ext_vector_type(8)))  float  v8f;

union FragBF { v16bf v; v8bf h[2]; };

__device__ __forceinline__ float silu_f(float x) {
    return x / (1.0f + __expf(-x));
}
__device__ __forceinline__ float softplus_f(float x) {
    return (x > 20.0f) ? x : log1pf(__expf(x));
}

// CDNA5 async copy: 16 bytes per lane, global -> LDS, tracked by ASYNCcnt.
__device__ __forceinline__ void async_b128(unsigned lds_addr, const void* gaddr) {
    asm volatile("global_load_async_to_lds_b128 %0, %1, off"
                 :: "v"(lds_addr), "v"(gaddr) : "memory");
}

// ---------------------------------------------------------------------------
// fp32 -> bf16 elementwise (weights)
// ---------------------------------------------------------------------------
__global__ __launch_bounds__(256)
void cvt_bf16_kernel(const float* __restrict__ in, __bf16* __restrict__ out,
                     size_t n) {
    size_t i = (size_t)blockIdx.x * blockDim.x + threadIdx.x;
    if (i < n) out[i] = (__bf16)in[i];
}

// ---------------------------------------------------------------------------
// LayerNorm over channel dim of x:(B, C, L) -> bf16 out:(B, L, C) row-major
// ---------------------------------------------------------------------------
__global__ __launch_bounds__(256)
void ln_kernel(const float* __restrict__ x, const float* __restrict__ g,
               const float* __restrict__ bt, __bf16* __restrict__ out,
               int C, int L) {
    int b  = blockIdx.y;
    int l0 = blockIdx.x * 64;
    int ll = threadIdx.x & 63;   // local l
    int cq = threadIdx.x >> 6;   // 0..3 channel slice
    int l  = l0 + ll;
    __shared__ float rs1[4][64], rs2[4][64], mu[64], rstd[64];
    float s1 = 0.f, s2 = 0.f;
    const float* xb = x + (size_t)b * C * L;
    if (l < L) {
        for (int c = cq; c < C; c += 4) {
            float v = xb[(size_t)c * L + l];    // coalesced along l
            s1 += v; s2 += v * v;
        }
    }
    rs1[cq][ll] = s1; rs2[cq][ll] = s2;
    __syncthreads();
    if (cq == 0) {
        float a = rs1[0][ll] + rs1[1][ll] + rs1[2][ll] + rs1[3][ll];
        float q = rs2[0][ll] + rs2[1][ll] + rs2[2][ll] + rs2[3][ll];
        float m = a / (float)C;
        mu[ll]   = m;
        rstd[ll] = rsqrtf(q / (float)C - m * m + 1e-5f);
    }
    __syncthreads();
    if (l < L) {
        float m = mu[ll], r = rstd[ll];
        __bf16* ob = out + ((size_t)b * L + l) * C;
        for (int c = cq; c < C; c += 4) {
            float v = xb[(size_t)c * L + l];
            ob[c] = (__bf16)((v - m) * r * g[c] + bt[c]);
        }
    }
}

// ---------------------------------------------------------------------------
// GEMM: Y[M,N] = X[M,K] * W[N,K]^T (+bias). X, W are bf16 in global memory.
// Block: 128 threads (4 waves). Tile 128(M) x 64(N); wave w owns rows
// [w*32, w*32+32) -> 8 WMMA per K-chunk (2 A-frags x 4 B-frags).
// K-chunks of 32, double-buffered LDS filled by async b128 copies
// (6 per wave per chunk; edge rows clamped so EXEC stays all-ones and the
// ASYNCcnt discipline stays exact). Last chunk peeled -> branchless loop.
// mode: 0 = plain(+bias), 1 = softplus(acc+bias), 2 = transposed store to
//       (B, N, Lp) layout with +bias (final output).
// Y (fp32) and Ybf (bf16 copy) each optional (nullptr to skip).
// ---------------------------------------------------------------------------
__global__ __launch_bounds__(128)
void gemm_bf16_wmma(const __bf16* __restrict__ X, int ldx,
                    const __bf16* __restrict__ W, int K,   // W: [N x K]
                    float* __restrict__ Y, int ldy,
                    __bf16* __restrict__ Ybf, int ldyb,
                    const float* __restrict__ bias,
                    int M, int N, int mode, int Lp) {
    // stride 40 elements (80B) per row: keeps b128 alignment, pads banks
    __shared__ __align__(16) __bf16 lsX[2][128 * 40];
    __shared__ __align__(16) __bf16 lsW[2][64 * 40];
    const int tid  = threadIdx.x;
    const int lane = tid & 31;
    const int wv   = tid >> 5;
    const int m0   = blockIdx.y * 128;
    const int n0   = blockIdx.x * 64;
    const int hi   = lane >> 4;
    const int l15  = lane & 15;

    const char* Xb = (const char*)X;
    const char* Wb = (const char*)W;

    // issue one 128x32 X tile + 64x32 W tile (bf16) into LDS buffer ib
    auto issue_tiles = [&](int k0, int ib) {
        unsigned baseX = (unsigned)(uintptr_t)(&lsX[ib][0]);
        unsigned baseW = (unsigned)(uintptr_t)(&lsW[ib][0]);
        #pragma unroll
        for (int i = 0; i < 4; ++i) {          // X: 8192B / (128 thr * 16B)
            int fb  = (i * 128 + tid) * 16;
            int row = fb >> 6;                 // 64B per row
            int cb  = fb & 63;
            int gm = m0 + row; if (gm > M - 1) gm = M - 1;   // clamp: EXEC full
            async_b128(baseX + row * 80 + cb,
                       Xb + ((size_t)gm * ldx + k0) * 2 + cb);
        }
        #pragma unroll
        for (int i = 0; i < 2; ++i) {          // W: 4096B
            int fb  = (i * 128 + tid) * 16;
            int row = fb >> 6;
            int cb  = fb & 63;
            int gn = n0 + row; if (gn > N - 1) gn = N - 1;
            async_b128(baseW + row * 80 + cb,
                       Wb + ((size_t)gn * K + k0) * 2 + cb);
        }
    };

    v8f acc[2][4] = {};

    // compute one K-chunk out of LDS buffer ib
    auto compute = [&](int ib) {
        // 16-bit A 16x32 layout (ISA 7.12.2): lane(0..15)=row M, hi selects
        // K halves {hi*8..hi*8+7, 16+hi*8..16+hi*8+7}
        FragBF ua[2];
        #pragma unroll
        for (int mt = 0; mt < 2; ++mt) {
            const __bf16* arow = &lsX[ib][(wv * 32 + mt * 16 + l15) * 40];
            ua[mt].h[0] = *(const v8bf*)(arow + hi * 8);
            ua[mt].h[1] = *(const v8bf*)(arow + 16 + hi * 8);
        }
        #pragma unroll
        for (int nt = 0; nt < 4; ++nt) {
            FragBF ub;   // B = W^T: same register pattern over row-major W[N,K]
            const __bf16* brow = &lsW[ib][(nt * 16 + l15) * 40];
            ub.h[0] = *(const v8bf*)(brow + hi * 8);
            ub.h[1] = *(const v8bf*)(brow + 16 + hi * 8);
            #pragma unroll
            for (int mt = 0; mt < 2; ++mt)
                acc[mt][nt] = __builtin_amdgcn_wmma_f32_16x16x32_bf16(
                    false, ua[mt].v, false, ub.v, (short)0, acc[mt][nt],
                    false, false);
        }
    };

    issue_tiles(0, 0);
    int ib = 0;
    int k0 = 0;
    for (; k0 + 32 < K; k0 += 32, ib ^= 1) {     // steady state: branchless
        issue_tiles(k0 + 32, ib ^ 1);
        // in-order async completion: <=6 outstanding => this chunk's 6 done;
        // barrier covers the other waves' copies.
        asm volatile("s_wait_asynccnt 0x6" ::: "memory");
        __syncthreads();
        compute(ib);
        __syncthreads();   // all waves done reading ib before refill
    }
    asm volatile("s_wait_asynccnt 0x0" ::: "memory");   // peeled last chunk
    __syncthreads();
    compute(ib);

    // C/D layout: lane 0-15 -> N=lane, M = vgpr r; lane 16-31 -> M = r+8
    #pragma unroll
    for (int mt = 0; mt < 2; ++mt) {
        const int mbase = m0 + wv * 32 + mt * 16 + hi * 8;
        #pragma unroll
        for (int nt = 0; nt < 4; ++nt) {
            int gn = n0 + nt * 16 + l15;
            if (gn >= N) continue;
            float bv = bias ? bias[gn] : 0.f;
            #pragma unroll
            for (int r = 0; r < 8; ++r) {
                int gm = mbase + r;
                if (gm >= M) continue;
                float v = acc[mt][nt][r] + bv;
                if (mode == 1) v = softplus_f(v);
                if (mode == 2) {
                    int bb = gm / Lp, l = gm % Lp;
                    Y[((size_t)bb * N + gn) * Lp + l] = v;
                } else {
                    if (Y)   Y[(size_t)gm * ldy + gn] = v;
                    if (Ybf) Ybf[(size_t)gm * ldyb + gn] = (__bf16)v;
                }
            }
        }
    }
}

// ---------------------------------------------------------------------------
// Depthwise conv1d along L (kernel 4) + SiLU; optional fp32 and bf16 outputs.
// ---------------------------------------------------------------------------
__global__ __launch_bounds__(256)
void dwconv_silu_kernel(const float* __restrict__ in, int ldin,
                        const float* __restrict__ w, const float* __restrict__ bias,
                        float* __restrict__ out32, __bf16* __restrict__ outbf,
                        int ldout, int C, int Lin, int Lout, int padL, int Bn) {
    size_t idx = (size_t)blockIdx.x * blockDim.x + threadIdx.x;
    size_t total = (size_t)Bn * Lout * C;
    if (idx >= total) return;
    int c = (int)(idx % C);
    int o = (int)((idx / C) % Lout);
    int b = (int)(idx / ((size_t)C * Lout));
    float acc = bias[c];
    #pragma unroll
    for (int k = 0; k < 4; ++k) {
        int li = o + k - padL;
        if (li >= 0 && li < Lin)
            acc += in[((size_t)b * Lin + li) * ldin + c] * w[c * 4 + k];
    }
    float v = silu_f(acc);
    size_t oidx = ((size_t)b * Lout + o) * ldout + c;
    if (out32) out32[oidx] = v;
    if (outbf) outbf[oidx] = (__bf16)v;
}

// ---------------------------------------------------------------------------
// Selective scan. One thread = one (b, d) channel; 16 states in registers.
// B_t/C_t staged through LDS in 64-step chunks.
// ---------------------------------------------------------------------------
__global__ __launch_bounds__(256)
void scan_kernel(const float* __restrict__ dt, const float* __restrict__ xdbl,
                 const float* __restrict__ u, const float* __restrict__ A_log,
                 float* __restrict__ y, int Lp) {
    const int D = 2048;
    int b = blockIdx.x >> 3;                       // 8 blocks per batch
    int d = ((blockIdx.x & 7) << 8) + threadIdx.x;
    float A[16], h[16];
    #pragma unroll
    for (int s = 0; s < 16; ++s) {
        A[s] = -__expf(A_log[d * 16 + s]);
        h[s] = 0.f;
    }
    __shared__ float Bs[64][16], Cs[64][16];
    for (int t0 = 0; t0 < Lp; t0 += 64) {
        int cn = min(64, Lp - t0);
        for (int i = threadIdx.x; i < 64 * 32; i += 256) {
            int tt = i >> 5, j = i & 31;
            if (tt < cn) {
                const float* row = xdbl + ((size_t)b * Lp + t0 + tt) * 96;
                if (j < 16) Bs[tt][j]      = row[64 + j];
                else        Cs[tt][j - 16] = row[80 + (j - 16)];
            }
        }
        __syncthreads();
        for (int tt = 0; tt < cn; ++tt) {
            size_t rr = ((size_t)b * Lp + t0 + tt) * D + d;
            float dtv = dt[rr], uv = u[rr];
            float yo = 0.f;
            #pragma unroll
            for (int s = 0; s < 16; ++s) {
                float dA = __expf(dtv * A[s]);
                h[s] = dA * h[s] + dtv * Bs[tt][s] * uv;
                yo  += h[s] * Cs[tt][s];
            }
            y[rr] = yo;
        }
        __syncthreads();
    }
}

// ---------------------------------------------------------------------------
// out_bf = (y + u*Dp) * silu(z)   (z = xz[:, 2048+c])
// ---------------------------------------------------------------------------
__global__ __launch_bounds__(256)
void combine_kernel(const float* __restrict__ y, const float* __restrict__ u,
                    const float* __restrict__ xz, const float* __restrict__ Dp,
                    __bf16* __restrict__ out, int C, size_t rows) {
    size_t idx = (size_t)blockIdx.x * blockDim.x + threadIdx.x;
    if (idx >= rows * (size_t)C) return;
    int c = (int)(idx % C);
    size_t r = idx / C;
    float z = xz[r * 4096 + 2048 + c];
    out[idx] = (__bf16)((y[idx] + u[idx] * Dp[c]) * silu_f(z));
}

// ---------------------------------------------------------------------------
extern "C" void kernel_launch(void* const* d_in, const int* in_sizes, int n_in,
                              void* d_out, int out_size, void* d_ws, size_t ws_size,
                              hipStream_t stream) {
    (void)in_sizes; (void)n_in; (void)out_size; (void)ws_size;
    const int B  = 2, L = 2048, Lp = 2049;
    const int DM = 1024, DI = 2048, DTR = 64;
    const size_t BL  = (size_t)B * L;    // 4096
    const size_t BLp = (size_t)B * Lp;   // 4098

    const float* x       = (const float*)d_in[0];
    const float* ln_g    = (const float*)d_in[1];
    const float* ln_b    = (const float*)d_in[2];
    const float* W_in    = (const float*)d_in[3];
    const float* b_in    = (const float*)d_in[4];
    const float* conv_w  = (const float*)d_in[5];
    const float* conv_b  = (const float*)d_in[6];
    const float* W_inprj = (const float*)d_in[7];
    const float* m_cw    = (const float*)d_in[8];
    const float* m_cb    = (const float*)d_in[9];
    const float* W_xprj  = (const float*)d_in[10];
    const float* W_dt    = (const float*)d_in[11];
    const float* b_dt    = (const float*)d_in[12];
    const float* A_log   = (const float*)d_in[13];
    const float* Dp      = (const float*)d_in[14];
    const float* W_oprj  = (const float*)d_in[15];
    const float* W_out   = (const float*)d_in[16];
    const float* b_out   = (const float*)d_in[17];
    float* out = (float*)d_out;

    // ---------------- workspace layout ----------------
    // fp32 region
    float* t2   = (float*)d_ws;              // BL  x 1024
    float* xz   = t2   + BL  * DM;           // BLp x 4096
    float* u    = xz   + BLp * 4096;         // BLp x 2048
    float* xdbl = u    + BLp * DI;           // BLp x 96
    float* dtb  = xdbl + BLp * 96;           // BLp x 2048
    float* ysc  = dtb  + BLp * DI;           // BLp x 2048
    // bf16 region (all sizes multiple of 8 elems -> 16B aligned)
    __bf16* hnorm_bf = (__bf16*)(ysc + BLp * DI);
    __bf16* u1_bf    = hnorm_bf + BL * DM;       // BLp x 1024
    __bf16* u_bf     = u1_bf  + BLp * DM;        // BLp x 2048
    __bf16* xdbl_bf  = u_bf   + BLp * DI;        // BLp x 96
    __bf16* yc_bf    = xdbl_bf + BLp * 96;       // BLp x 2048
    __bf16* t3_bf    = yc_bf  + BLp * DI;        // BLp x 1024
    __bf16* Wb_in    = t3_bf  + BLp * DM;
    __bf16* Wb_inprj = Wb_in    + (size_t)DM * DM;
    __bf16* Wb_xprj  = Wb_inprj + (size_t)4096 * DM;
    __bf16* Wb_dt    = Wb_xprj  + (size_t)96 * DI;
    __bf16* Wb_oprj  = Wb_dt    + (size_t)DI * DTR;
    __bf16* Wb_out   = Wb_oprj  + (size_t)DM * DI;

    auto cvt = [&](const float* src, __bf16* dst, size_t n) {
        cvt_bf16_kernel<<<(int)((n + 255) / 256), 256, 0, stream>>>(src, dst, n);
    };
    cvt(W_in,    Wb_in,    (size_t)DM * DM);
    cvt(W_inprj, Wb_inprj, (size_t)4096 * DM);
    cvt(W_xprj,  Wb_xprj,  (size_t)96 * DI);
    cvt(W_dt,    Wb_dt,    (size_t)DI * DTR);
    cvt(W_oprj,  Wb_oprj,  (size_t)DM * DI);
    cvt(W_out,   Wb_out,   (size_t)DM * DM);

    auto gemm = [&](const __bf16* X, int ldx, const __bf16* W, int K,
                    float* Y, int ldy, __bf16* Ybf, int ldyb,
                    const float* bias, int M, int N, int mode) {
        dim3 g((N + 63) / 64, (M + 127) / 128);
        gemm_bf16_wmma<<<g, 128, 0, stream>>>(X, ldx, W, K, Y, ldy, Ybf, ldyb,
                                              bias, M, N, mode, Lp);
    };

    // 1) layernorm (B,C,L) -> bf16 (B,L,C)
    ln_kernel<<<dim3((L + 63) / 64, B), 256, 0, stream>>>(x, ln_g, ln_b,
                                                          hnorm_bf, DM, L);
    // 2) h @ W_in^T + b_in -> t2 (fp32, conv input)
    gemm(hnorm_bf, DM, Wb_in, DM, t2, DM, nullptr, 0, b_in, (int)BL, DM, 0);
    // 3) dwconv pad(2,2) + silu -> u1 (bf16 only, length Lp)
    {
        size_t tot = (size_t)B * Lp * DM;
        dwconv_silu_kernel<<<(int)((tot + 255) / 256), 256, 0, stream>>>(
            t2, DM, conv_w, conv_b, nullptr, u1_bf, DM, DM, L, Lp, 2, B);
    }
    // 4) u1 @ W_inproj^T -> xz (fp32: xi for conv2, z for gating)
    gemm(u1_bf, DM, Wb_inprj, DM, xz, 4096, nullptr, 0, nullptr, (int)BLp, 4096, 0);
    // 5) causal dwconv pad(3,0) + silu on xi -> u (fp32 + bf16)
    {
        size_t tot = (size_t)B * Lp * DI;
        dwconv_silu_kernel<<<(int)((tot + 255) / 256), 256, 0, stream>>>(
            xz, 4096, m_cw, m_cb, u, u_bf, DI, DI, Lp, Lp, 3, B);
    }
    // 6) u @ W_xproj^T -> x_dbl (fp32 for scan B/C, bf16 for dt GEMM)
    gemm(u_bf, DI, Wb_xprj, DI, xdbl, 96, xdbl_bf, 96, nullptr, (int)BLp, 96, 0);
    // 7) softplus(dt @ W_dt^T + b_dt) -> dtb (fp32)
    gemm(xdbl_bf, 96, Wb_dt, DTR, dtb, DI, nullptr, 0, b_dt, (int)BLp, DI, 1);
    // 8) selective scan -> ysc (fp32)
    scan_kernel<<<B * 8, 256, 0, stream>>>(dtb, xdbl, u, A_log, ysc, Lp);
    // 9) gating: yc = (y + u*Dp) * silu(z) -> bf16
    {
        size_t tot = (size_t)B * Lp * DI;
        combine_kernel<<<(int)((tot + 255) / 256), 256, 0, stream>>>(
            ysc, u, xz, Dp, yc_bf, DI, BLp);
    }
    // 10) yc @ W_outproj^T -> t3 (bf16 only)
    gemm(yc_bf, DI, Wb_oprj, DI, nullptr, 0, t3_bf, DM, nullptr, (int)BLp, DM, 0);
    // 11) (@ W_out^T + b_out) transposed store -> (B, DM, Lp)
    gemm(t3_bf, DM, Wb_out, DM, out, 0, nullptr, 0, b_out, (int)BLp, DM, 2);
}
